// VA_subnetwork_80633716015532
// MI455X (gfx1250) — compile-verified, matrix operand order checked by reading, not run on hardware
//
#include <hip/hip_runtime.h>

#define Bsz   1024
#define Tlen  300
#define Jn    25
#define Hd    256
#define Iin   75
#define KPAD  96
#define BT    16
#define THREADS 256
#define KT_X  3
#define KT_H  8

typedef __attribute__((ext_vector_type(16))) _Float16 v16h;
typedef __attribute__((ext_vector_type(8)))  _Float16 v8h;
typedef __attribute__((ext_vector_type(8)))  float    v8f;
typedef __attribute__((ext_vector_type(4)))  unsigned int v4u;
typedef __attribute__((ext_vector_type(8)))  int      v8i;
typedef __attribute__((ext_vector_type(4)))  int      v4i;

// K index inside a 16x32 (A) / 32x16 (B) 16-bit WMMA fragment:
// lane-group g (lane>=16), element e (0..15 per lane).
__device__ __forceinline__ int frag_k(int e, int g) {
  int d = e >> 1;
  return ((d & 4) ? 16 : 0) + g * 8 + (d & 3) * 2 + (e & 1);
}

// ---------------------------------------------------------------------------
// TDM: async 2D tile load (16 rows x 75 f32, row stride Tlen*75) into LDS.
// D# groups built per CDNA5 ISA 8.3-8.6; 2D tensor -> groups 2/3 unused.
// ---------------------------------------------------------------------------
__device__ __forceinline__ void tdm_load_x(const float* gsrc, unsigned lds_off) {
  unsigned long long ga = (unsigned long long)(uintptr_t)gsrc;
  v4u g0;
  g0[0] = 1u;                                              // count=1 (valid)
  g0[1] = lds_off;                                         // lds_addr (bytes)
  g0[2] = (unsigned)(ga & 0xffffffffu);                    // global_addr lo
  g0[3] = (unsigned)((ga >> 32) & 0x01ffffffu) | (2u << 30); // addr hi | type=2
  v8i g1;
  g1[0] = (2 << 16);                 // wg_mask=0, data_size=2 (4 bytes)
  g1[1] = (int)((unsigned)Iin << 16);          // tensor_dim0[15:0] @ [31:16]
  g1[2] = (int)((unsigned)BT << 16);           // dim0 hi=0 | tensor_dim1 lo
  g1[3] = (int)((unsigned)Iin << 16);          // dim1 hi=0 | tile_dim0=75
  g1[4] = BT;                                  // tile_dim1=16, tile_dim2=0
  g1[5] = Tlen * Iin;                          // tensor_dim0_stride lo32
  g1[6] = 0;                                   // stride hi | dim1_stride lo
  g1[7] = 0;
  v4i z4 = {0, 0, 0, 0};
#if __clang_major__ >= 23
  v8i z8 = {0, 0, 0, 0, 0, 0, 0, 0};
  __builtin_amdgcn_tensor_load_to_lds(g0, g1, z4, z4, z8, 0);
#else
  __builtin_amdgcn_tensor_load_to_lds(g0, g1, z4, z4, 0);
#endif
}

// ---------------------------------------------------------------------------
// Prep: pack Wih (768x75 -> Kpad 96) and Whh (768x256) of both GRUs into f16
// B-fragment tiles (tile = 32x16, 512 f16; lane L reads 16 contiguous f16 at
// L*16).  Fold biases: brz = bih+bhh (r,z), bin = bih_n, bhn = bhh_n.
// Tile regions (512-elem tiles): rtWih[0,144) dtWih[144,288) rtWhh[288,672)
// dtWhh[672,1056); within region tile (nt,kt) at nt*KT+kt.
// ---------------------------------------------------------------------------
__global__ void prep_kernel(const float* __restrict__ rtWih, const float* __restrict__ rtWhh,
                            const float* __restrict__ rtBih, const float* __restrict__ rtBhh,
                            const float* __restrict__ dtWih, const float* __restrict__ dtWhh,
                            const float* __restrict__ dtBih, const float* __restrict__ dtBhh,
                            _Float16* __restrict__ wtiles, float* __restrict__ biasbuf)
{
  int bid = blockIdx.x;
  int tid = threadIdx.x;
  if (bid < 1056) {
    const float* W; int KT, Ksrc, tr;
    if (bid < 144)      { W = rtWih; KT = KT_X; Ksrc = Iin; tr = bid;       }
    else if (bid < 288) { W = dtWih; KT = KT_X; Ksrc = Iin; tr = bid - 144; }
    else if (bid < 672) { W = rtWhh; KT = KT_H; Ksrc = Hd;  tr = bid - 288; }
    else                { W = dtWhh; KT = KT_H; Ksrc = Hd;  tr = bid - 672; }
    int nt = tr / KT, kt = tr % KT;
    _Float16* dst = wtiles + (size_t)bid * 512;
    for (int ei = tid; ei < 512; ei += THREADS) {
      int lane = ei >> 4, e = ei & 15;
      int g = lane >> 4, nl = lane & 15;
      int K = kt * 32 + frag_k(e, g);
      int N = nt * 16 + nl;
      float v = (K < Ksrc) ? W[(size_t)N * Ksrc + K] : 0.0f;   // B[K][N]=W[N][K]
      dst[ei] = (_Float16)v;
    }
  } else {
    for (int i = tid; i < 2048; i += THREADS) {
      int g = i >> 10, r = i & 1023;
      const float* bih = g ? dtBih : rtBih;
      const float* bhh = g ? dtBhh : rtBhh;
      float v;
      if (r < 512)      v = bih[r] + bhh[r];
      else if (r < 768) v = bih[512 + (r - 512)];
      else              v = bhh[512 + (r - 768)];
      biasbuf[i] = v;
    }
  }
}

// ---------------------------------------------------------------------------
// Persistent GRU kernel: one workgroup = 16 batch rows, loops all T steps.
// 8 waves: waves 0-3 -> rt GRU, waves 4-7 -> dt GRU; each wave owns 16 of the
// 64 gate column-tiles of its GRU (32 rz fused, 16 i_n, 16 h_n).
// x tiles double-buffered in LDS via TDM (issued one step ahead).
// ---------------------------------------------------------------------------
__global__ void gru_kernel(const float* __restrict__ x,
                           const float* __restrict__ rt_h0,
                           const float* __restrict__ dt_h0,
                           const float* __restrict__ rt_fc,
                           const float* __restrict__ dt_fc,
                           const _Float16* __restrict__ wtiles,
                           const float* __restrict__ biasbuf,
                           float* __restrict__ out)
{
  extern __shared__ char smem[];
  float*    hF  = (float*)(smem);              // [2][16][256] f32 hidden
  float*    grz = (float*)(smem + 32768);      // [2][16][512] sigmoid(r|z)
  float*    gin = (float*)(smem + 98304);      // [2][16][256] i_n
  float*    ghn = (float*)(smem + 131072);     // [2][16][256] h_n
  float*    xg  = (float*)(smem + 163840);     // [2][16][75]  x f32 (TDM dst)
  float*    fco = (float*)(smem + 173440);     // [2][16][4]   fc outputs
  float*    Rm  = (float*)(smem + 173952);     // [16][12]     rotation mats
  _Float16* hB  = (_Float16*)(smem + 174720);  // [2][16][256] f16 hidden
  _Float16* xbf = (_Float16*)(smem + 191104);  // [16][96]     f16 x (padded)

  const int tid  = threadIdx.x;
  const int lane = tid & 31;
  const int wave = tid >> 5;
  const int gru  = wave >> 2;   // 0 = rt, 1 = dt
  const int wsub = wave & 3;
  const int grp  = lane >> 4;
  const int nl   = lane & 15;
  const int b0   = blockIdx.x * BT;

  const unsigned xg_off = (unsigned)(unsigned long long)(uintptr_t)(void*)xg;

  // kick off TDM for step 0 into buffer 0
  if (wave == 0) {
    tdm_load_x(x + (size_t)b0 * Tlen * Iin, xg_off);
  }

  // initial hidden state
  for (int i = tid; i < 2 * BT * Hd; i += THREADS) {
    int g = i / (BT * Hd); int r = i % (BT * Hd);
    int m = r / Hd, k = r % Hd;
    float v = (g ? dt_h0 : rt_h0)[(size_t)(b0 + m) * Hd + k];
    hF[i] = v;
    hB[i] = (_Float16)v;
  }
  __syncthreads();

  const _Float16* wihLane = wtiles + (size_t)(gru ? 144 : 0) * 512 + lane * 16;
  const _Float16* whhLane = wtiles + (size_t)(288 + gru * 384) * 512 + lane * 16;
  const float*    bias    = biasbuf + gru * 1024;

  for (int t = 0; t < Tlen; ++t) {
    const float* xcur = xg + (t & 1) * (BT * Iin);

    // ---- phase 0: wait for this step's TDM tile, launch next, convert ----
    if (wave == 0) {
      __builtin_amdgcn_s_wait_tensorcnt(0);
      if (t + 1 < Tlen) {
        tdm_load_x(x + ((size_t)b0 * Tlen + (t + 1)) * Iin,
                   xg_off + (unsigned)(((t + 1) & 1) * (BT * Iin * 4)));
      }
    }
    __syncthreads();
    for (int i = tid; i < BT * KPAD; i += THREADS) {
      int m = i / KPAD, c = i % KPAD;
      float v = (c < Iin) ? xcur[m * Iin + c] : 0.0f;
      xbf[m * KPAD + c] = (_Float16)v;
    }
    __syncthreads();

    // ---- phase 1: WMMA gate GEMMs ----
    // A fragment = two contiguous 8-elem runs per lane: K = g*8.. and 16+g*8..
    v16h xa[KT_X];
#pragma unroll
    for (int kt = 0; kt < KT_X; ++kt) {
      v8h lo = *(const v8h*)(xbf + nl * KPAD + kt * 32 + grp * 8);
      v8h hi = *(const v8h*)(xbf + nl * KPAD + kt * 32 + 16 + grp * 8);
      xa[kt] = __builtin_shufflevector(lo, hi, 0, 1, 2, 3, 4, 5, 6, 7,
                                       8, 9, 10, 11, 12, 13, 14, 15);
    }
    v16h ha[KT_H];
    const _Float16* hrow = hB + gru * BT * Hd;
#pragma unroll
    for (int kt = 0; kt < KT_H; ++kt) {
      v8h lo = *(const v8h*)(hrow + nl * Hd + kt * 32 + grp * 8);
      v8h hi = *(const v8h*)(hrow + nl * Hd + kt * 32 + 16 + grp * 8);
      ha[kt] = __builtin_shufflevector(lo, hi, 0, 1, 2, 3, 4, 5, 6, 7,
                                       8, 9, 10, 11, 12, 13, 14, 15);
    }

    for (int i = 0; i < 16; ++i) {
      int tl = wsub * 16 + i;
      int kind, j;                       // 0: rz fused, 1: i_n, 2: h_n
      if (tl < 32)      { kind = 0; j = tl;      }
      else if (tl < 48) { kind = 1; j = tl - 32; }
      else              { kind = 2; j = tl - 48; }
      int ntB = (kind == 0) ? j : (32 + j);

      float bv = (kind == 0) ? bias[j * 16 + nl]
               : (kind == 1) ? bias[512 + j * 16 + nl]
                             : bias[768 + j * 16 + nl];
      v8f acc;
#pragma unroll
      for (int r = 0; r < 8; ++r) acc[r] = bv;

      if (kind != 2) {
#pragma unroll
        for (int kt = 0; kt < KT_X; ++kt) {
          v16h bf = *(const v16h*)(wihLane + ((size_t)(ntB * KT_X + kt)) * 512);
          acc = __builtin_amdgcn_wmma_f32_16x16x32_f16(false, xa[kt], false, bf,
                                                       (short)0, acc, false, false);
        }
      }
      if (kind != 1) {
#pragma unroll
        for (int kt = 0; kt < KT_H; ++kt) {
          v16h bf = *(const v16h*)(whhLane + ((size_t)(ntB * KT_H + kt)) * 512);
          acc = __builtin_amdgcn_wmma_f32_16x16x32_f16(false, ha[kt], false, bf,
                                                       (short)0, acc, false, false);
        }
      }
#pragma unroll
      for (int r = 0; r < 8; ++r) {
        int m = r + grp * 8;
        float v = acc[r];
        if (kind == 0) {
          v = 1.0f / (1.0f + __expf(-v));                 // sigmoid(r|z)
          grz[(gru * BT + m) * 512 + j * 16 + nl] = v;
        } else if (kind == 1) {
          gin[(gru * BT + m) * Hd + j * 16 + nl] = v;
        } else {
          ghn[(gru * BT + m) * Hd + j * 16 + nl] = v;
        }
      }
    }
    __syncthreads();

    // ---- phase 2: GRU elementwise update ----
    for (int i = tid; i < 2 * BT * Hd; i += THREADS) {
      int g = i / (BT * Hd); int rr = i % (BT * Hd);
      int m = rr / Hd, c = rr % Hd;
      float rg = grz[(g * BT + m) * 512 + c];
      float zg = grz[(g * BT + m) * 512 + 256 + c];
      float n  = tanhf(gin[i] + rg * ghn[i]);
      float hnew = (1.0f - zg) * n + zg * hF[i];
      hF[i] = hnew;
      hB[i] = (_Float16)hnew;
    }
    __syncthreads();

    // ---- phase 3a: tiny FC heads (rt, dt) ----
    if (tid < 96) {
      int g = tid / 48, rr = tid % 48;
      int m = rr / 3, j = rr % 3;
      const float* W  = (g ? dt_fc : rt_fc) + j * Hd;
      const float* hv = hF + (g * BT + m) * Hd;
      float s = 0.0f;
      for (int k = 0; k < Hd; ++k) s += hv[k] * W[k];
      fco[(g * BT + m) * 4 + j] = s;
    }
    __syncthreads();

    // ---- phase 3b: rotation matrices ----
    if (tid < BT) {
      int m = tid;
      float a  = fco[m * 4 + 0], b = fco[m * 4 + 1], gg = fco[m * 4 + 2];
      float ca = cosf(a),  sa = sinf(a);
      float cb = cosf(b),  sb = sinf(b);
      float cg = cosf(gg), sg = sinf(gg);
      float* R = Rm + m * 12;
      R[0] = cb * cg;                 R[1] = -cb * sg;                 R[2] = -sb;
      R[3] = sa * sb * cg + ca * sg;  R[4] = ca * cg - sa * sb * sg;   R[5] = sa * cb;
      R[6] = ca * sb * cg - sa * sg;  R[7] = -ca * sb * sg - sa * cg;  R[8] = ca * cb;
    }
    __syncthreads();

    // ---- phase 3c: y = swap(R @ x3) + dt ----
    for (int p = tid; p < BT * Jn; p += THREADS) {
      int m = p / Jn, jt = p % Jn;
      const float* R = Rm + m * 12;
      float x0 = xcur[m * Iin + jt];
      float x1 = xcur[m * Iin + 25 + jt];
      float x2 = xcur[m * Iin + 50 + jt];
      float d0 = fco[(BT + m) * 4 + 0];
      float d1 = fco[(BT + m) * 4 + 1];
      float d2 = fco[(BT + m) * 4 + 2];
      size_t o = (((size_t)(b0 + m) * Tlen + t) * Jn + jt) * 3;
      out[o + 0] = R[0] * x0 + R[1] * x1 + R[2] * x2 + d0;
      out[o + 1] = R[3] * x0 + R[4] * x1 + R[5] * x2 + d1;
      out[o + 2] = R[6] * x0 + R[7] * x1 + R[8] * x2 + d2;
    }
    __syncthreads();
  }
}

extern "C" void kernel_launch(void* const* d_in, const int* in_sizes, int n_in,
                              void* d_out, int out_size, void* d_ws, size_t ws_size,
                              hipStream_t stream) {
  const float* x     = (const float*)d_in[0];
  const float* rt_h  = (const float*)d_in[1];
  const float* dt_h  = (const float*)d_in[2];
  const float* rtWih = (const float*)d_in[3];
  const float* rtWhh = (const float*)d_in[4];
  const float* rtBih = (const float*)d_in[5];
  const float* rtBhh = (const float*)d_in[6];
  const float* dtWih = (const float*)d_in[7];
  const float* dtWhh = (const float*)d_in[8];
  const float* dtBih = (const float*)d_in[9];
  const float* dtBhh = (const float*)d_in[10];
  const float* rtFc  = (const float*)d_in[11];
  const float* dtFc  = (const float*)d_in[12];

  _Float16* wt   = (_Float16*)d_ws;
  float*    bias = (float*)((char*)d_ws + (size_t)1056 * 512 * sizeof(_Float16));

  prep_kernel<<<1057, THREADS, 0, stream>>>(rtWih, rtWhh, rtBih, rtBhh,
                                            dtWih, dtWhh, dtBih, dtBhh, wt, bias);

  const size_t smem = 194176;  // ~190 KB dynamic LDS (WGP supports 320 KB)
  gru_kernel<<<Bsz / BT, THREADS, smem, stream>>>(x, rt_h, dt_h, rtFc, dtFc,
                                                  wt, bias, (float*)d_out);
}
